// TransformerMoEBlock_24043226923899
// MI455X (gfx1250) — matrix operand order repaired and checked by reading, not compiled, not run on hardware
//
#include <hip/hip_runtime.h>

// ============================================================================
// TransformerMoE block for gfx1250 (MI455X), wave32 + WMMA bf16.
//
// Pipeline (all on `stream`):
//  1. rmsnorm(x_f32)            -> h  (bf16)
//  2. transpose+cvt all weights -> bf16 "BT" layout (BT[n][k]); W1/W2 bf16
//     (134 MB) then live in the 192MB L2 for the expert GEMMs.
//  3. q = (h@Wq)*HD^-0.5 (bf16), k = h@Wk (bf16), vT = (h@Wv)^T (bf16)
//  4. flash attention per (b, head, 64 q rows): WMMA scores, online softmax,
//     P staged via LDS into A-frag layout, WMMA P@V            -> o (bf16)
//  5. x1 = o@Wo + h (bf16)
//  6. h2 = rmsnorm(x1) (bf16)
//  7. gating: logits=h2@Wg (f32 VALU, tiny), top-2, renorm softmax,
//     atomic counts -> scan -> row assignment (grouped-GEMM routing)
//  8. per-expert: act = gelu(h2_gathered @ W1[e]) (bf16, WMMA)
//  9. per-expert: C = act @ W2[e] (f32, WMMA)
// 10. out = x_orig + g0*C[row0] + g1*C[row1]
// ============================================================================

typedef __bf16 bf16_t;
typedef __attribute__((ext_vector_type(16))) __bf16 v16bf;
typedef __attribute__((ext_vector_type(8)))  __bf16 v8bf;
typedef __attribute__((ext_vector_type(8)))  float  v8f;

#define N_TOK 4096   // B*S
#define DIM   1024
#define HDIM  64
#define NHEAD 16
#define SEQ   2048
#define NEXP  8
#define HID   4096
#define RMAX  8192   // N_TOK * K assignments

__device__ __forceinline__ int imin(int a, int b) { return a < b ? a : b; }

// ---- WMMA fragment loaders (CDNA5 16-bit layouts, ISA 7.12.2) -------------
// A 16x32: lanes 0-15 -> row (lane&15), K chunks [0..7] & [16..23];
//          lanes 16-31 -> same rows, K chunks [8..15] & [24..31].
__device__ __forceinline__ v16bf frag_a(const bf16_t* __restrict__ base,
                                        int lda, int row, int k, int lane) {
  const bf16_t* p = base + (size_t)row * lda + k + ((lane & 16) ? 8 : 0);
  v8bf lo = *(const v8bf*)p;
  v8bf hi = *(const v8bf*)(p + 16);
  v16bf r;
#pragma unroll
  for (int i = 0; i < 8; ++i) { r[i] = lo[i]; r[i + 8] = hi[i]; }
  return r;
}

// B 32x16 from BT (row-major [N x K]): lane (n&15) holds column n;
// lanes 0-15 -> K [0..7] & [16..23], lanes 16-31 -> K [8..15] & [24..31].
__device__ __forceinline__ v16bf frag_b(const bf16_t* __restrict__ bt,
                                        int ldbt, int col, int k, int lane) {
  const bf16_t* p = bt + (size_t)col * ldbt + k + ((lane & 16) ? 8 : 0);
  v8bf lo = *(const v8bf*)p;
  v8bf hi = *(const v8bf*)(p + 16);
  v16bf r;
#pragma unroll
  for (int i = 0; i < 8; ++i) { r[i] = lo[i]; r[i + 8] = hi[i]; }
  return r;
}

__device__ __forceinline__ v8f wmma_bf16(v16bf a, v16bf b, v8f c) {
  return __builtin_amdgcn_wmma_f32_16x16x32_bf16(false, a, false, b,
                                                 (short)0, c, false, false);
}

__device__ __forceinline__ float gelu_f(float x) {
  float x3 = x * x * x;
  float t  = tanhf(0.7978845608028654f * (x + 0.044715f * x3));
  return 0.5f * x * (1.0f + t);
}

// ---- RMSNorm (zeta SimpleRMSNorm): x / max(||x||,eps) * sqrt(D) -----------
template <typename T>
__global__ __launch_bounds__(256) void rmsnorm_kernel(const T* __restrict__ in,
                                                      bf16_t* __restrict__ out) {
  __shared__ float red[8];
  int t = blockIdx.x;
  const T* row = in + (size_t)t * DIM;
  float vals[4];
  float ss = 0.f;
#pragma unroll
  for (int i = 0; i < 4; ++i) {
    float v = (float)row[threadIdx.x + i * 256];
    vals[i] = v; ss += v * v;
  }
  for (int x = 1; x < 32; x <<= 1) ss += __shfl_xor(ss, x, 32);
  int lane = threadIdx.x & 31, w = threadIdx.x >> 5;
  if (lane == 0) red[w] = ss;
  __syncthreads();
  if (w == 0) {
    float v = red[lane & 7];
    for (int x = 1; x < 8; x <<= 1) v += __shfl_xor(v, x, 32);
    if (lane == 0) red[0] = v;
  }
  __syncthreads();
  float sc = 32.0f / fmaxf(sqrtf(red[0]), 1e-12f);  // sqrt(1024)=32
  bf16_t* orow = out + (size_t)t * DIM;
#pragma unroll
  for (int i = 0; i < 4; ++i)
    orow[threadIdx.x + i * 256] = (bf16_t)(vals[i] * sc);
}

// ---- Transpose + f32->bf16 convert: in [R x C] -> out [C x R] -------------
__global__ __launch_bounds__(256) void transpose_cvt_kernel(
    const float* __restrict__ in, bf16_t* __restrict__ out, int R, int C) {
  __shared__ float tile[32][33];
  size_t bofs = (size_t)blockIdx.z * R * C;
  const float* src = in + bofs;
  bf16_t* dst = out + bofs;
  int c0 = blockIdx.x * 32, r0 = blockIdx.y * 32;
  int tx = threadIdx.x & 31, ty = threadIdx.x >> 5;
#pragma unroll
  for (int i = 0; i < 4; ++i) {
    int r = r0 + ty + i * 8, c = c0 + tx;
    tile[ty + i * 8][tx] = (r < R && c < C) ? src[(size_t)r * C + c] : 0.f;
  }
  __syncthreads();
#pragma unroll
  for (int i = 0; i < 4; ++i) {
    int c = c0 + ty + i * 8, r = r0 + tx;
    if (c < C && r < R) dst[(size_t)c * R + r] = (bf16_t)tile[tx][ty + i * 8];
  }
}

// ---- Generic WMMA GEMM: C[MxN] = A[MxK] * BT[NxK]^T, various epilogues ----
enum { EPI_F32 = 0, EPI_BF16 = 1, EPI_SCALE_BF16 = 2, EPI_ADD_BF16 = 3,
       EPI_TRANS_BF16 = 4 };

template <int EPI>
__global__ __launch_bounds__(256) void gemm_bf16_wmma(
    const bf16_t* __restrict__ A, int lda,
    const bf16_t* __restrict__ BT, int ldbt,
    void* __restrict__ Cout, int ldc,
    int M, int N, int Kdim,
    const bf16_t* __restrict__ resid, float scale) {
  int tid = threadIdx.x, lane = tid & 31, wave = tid >> 5;
  int wm = wave & 3, wn = wave >> 2;                 // 4x2 waves per block
  int m0 = blockIdx.y * 64 + wm * 16;
  int n0 = blockIdx.x * 128 + wn * 64;
  if (m0 >= M || n0 >= N) return;                    // wave-uniform
  int rowA = imin(m0 + (lane & 15), M - 1);
  int colB[4];
#pragma unroll
  for (int j = 0; j < 4; ++j) colB[j] = imin(n0 + j * 16 + (lane & 15), N - 1);
  v8f acc[4] = {};
  for (int k = 0; k < Kdim; k += 32) {
    v16bf a = frag_a(A, lda, rowA, k, lane);
#pragma unroll
    for (int j = 0; j < 4; ++j) {
      v16bf b = frag_b(BT, ldbt, colB[j], k, lane);
      acc[j] = wmma_bf16(a, b, acc[j]);
    }
  }
#pragma unroll
  for (int j = 0; j < 4; ++j) {
    int col = n0 + j * 16 + (lane & 15);
    if (col >= N) continue;
#pragma unroll
    for (int r = 0; r < 8; ++r) {
      int row = m0 + r + ((lane & 16) ? 8 : 0);
      if (row >= M) continue;
      float v = acc[j][r];
      if (EPI == EPI_F32)
        ((float*)Cout)[(size_t)row * ldc + col] = v;
      else if (EPI == EPI_BF16)
        ((bf16_t*)Cout)[(size_t)row * ldc + col] = (bf16_t)v;
      else if (EPI == EPI_SCALE_BF16)
        ((bf16_t*)Cout)[(size_t)row * ldc + col] = (bf16_t)(v * scale);
      else if (EPI == EPI_ADD_BF16)
        ((bf16_t*)Cout)[(size_t)row * ldc + col] =
            (bf16_t)(v + (float)resid[(size_t)row * ldc + col]);
      else if (EPI == EPI_TRANS_BF16)
        ((bf16_t*)Cout)[(size_t)col * ldc + row] = (bf16_t)v;
    }
  }
}

// ---- Flash attention, one (b, head, 64 q-rows) per block (4 waves) --------
__global__ __launch_bounds__(128) void attn_kernel(
    const bf16_t* __restrict__ q,     // [N_TOK x DIM], pre-scaled by HD^-0.5
    const bf16_t* __restrict__ kmat,  // [N_TOK x HDIM]
    const bf16_t* __restrict__ vT,    // [HDIM x N_TOK]
    bf16_t* __restrict__ o) {         // [N_TOK x DIM]
  __shared__ bf16_t pbuf[4][16][64];
  int tid = threadIdx.x, lane = tid & 31, wave = tid >> 5;
  int b = blockIdx.z, hh = blockIdx.y;
  int qtok = b * SEQ + blockIdx.x * 64 + wave * 16;
  v8f o_acc[4] = {};
  float mrow[8], lrow[8];
#pragma unroll
  for (int r = 0; r < 8; ++r) { mrow[r] = -3.0e38f; lrow[r] = 0.f; }
  int arow = qtok + (lane & 15);
  v16bf aq0 = frag_a(q, DIM, arow, hh * HDIM, lane);
  v16bf aq1 = frag_a(q, DIM, arow, hh * HDIM + 32, lane);

  for (int kb = 0; kb < SEQ; kb += 64) {
    v8f s_acc[4] = {};
#pragma unroll
    for (int j = 0; j < 4; ++j) {
      int keytok = b * SEQ + kb + j * 16 + (lane & 15);
      s_acc[j] = wmma_bf16(aq0, frag_b(kmat, HDIM, keytok, 0, lane), s_acc[j]);
      s_acc[j] = wmma_bf16(aq1, frag_b(kmat, HDIM, keytok, 32, lane), s_acc[j]);
    }
    // online softmax; C layout: rows r(+8 for upper half) across 16 col-lanes
#pragma unroll
    for (int r = 0; r < 8; ++r) {
      float m = s_acc[0][r];
#pragma unroll
      for (int j = 1; j < 4; ++j) m = fmaxf(m, s_acc[j][r]);
      for (int x = 1; x < 16; x <<= 1) m = fmaxf(m, __shfl_xor(m, x, 32));
      float mn = fmaxf(mrow[r], m);
      float corr = __expf(mrow[r] - mn);
      mrow[r] = mn;
      lrow[r] *= corr;
#pragma unroll
      for (int j = 0; j < 4; ++j) o_acc[j][r] *= corr;
      float s = 0.f;
#pragma unroll
      for (int j = 0; j < 4; ++j) {
        float p = __expf(s_acc[j][r] - mn);
        s_acc[j][r] = p; s += p;
      }
      for (int x = 1; x < 16; x <<= 1) s += __shfl_xor(s, x, 32);
      lrow[r] += s;
    }
    // stage P (bf16) through LDS to re-layout C-frag -> A-frag
#pragma unroll
    for (int j = 0; j < 4; ++j)
#pragma unroll
      for (int r = 0; r < 8; ++r)
        pbuf[wave][r + ((lane & 16) ? 8 : 0)][j * 16 + (lane & 15)] =
            (bf16_t)s_acc[j][r];
    asm volatile("s_wait_dscnt 0" ::: "memory");  // same-wave LDS RAW
    const bf16_t* pb = &pbuf[wave][0][0];
#pragma unroll
    for (int s = 0; s < 2; ++s) {
      v16bf ap = frag_a(pb, 64, lane & 15, s * 32, lane);
#pragma unroll
      for (int j = 0; j < 4; ++j) {
        v16bf bv = frag_b(vT, N_TOK, j * 16 + (lane & 15),
                          b * SEQ + kb + s * 32, lane);
        o_acc[j] = wmma_bf16(ap, bv, o_acc[j]);
      }
    }
  }
#pragma unroll
  for (int j = 0; j < 4; ++j) {
    int col = hh * HDIM + j * 16 + (lane & 15);
#pragma unroll
    for (int r = 0; r < 8; ++r) {
      int row = qtok + r + ((lane & 16) ? 8 : 0);
      o[(size_t)row * DIM + col] = (bf16_t)(o_acc[j][r] / lrow[r]);
    }
  }
}

// ---- Gating: logits = h2@Wg, top-2, renormalized softmax, count experts ---
__global__ __launch_bounds__(256) void gate_topk_kernel(
    const bf16_t* __restrict__ h2, const float* __restrict__ Wg,
    int* __restrict__ topi, float* __restrict__ gates,
    int* __restrict__ counts) {
  int lane = threadIdx.x & 31, w = threadIdx.x >> 5;
  int t = blockIdx.x * 8 + w;
  const bf16_t* row = h2 + (size_t)t * DIM;
  float acc[NEXP] = {};
  for (int d = lane; d < DIM; d += 32) {
    float xv = (float)row[d];
    const float* wg = Wg + (size_t)d * NEXP;
#pragma unroll
    for (int e = 0; e < NEXP; ++e) acc[e] += xv * wg[e];
  }
#pragma unroll
  for (int e = 0; e < NEXP; ++e)
    for (int x = 1; x < 32; x <<= 1) acc[e] += __shfl_xor(acc[e], x, 32);
  if (lane == 0) {
    int i0 = 0; float v0 = acc[0];
#pragma unroll
    for (int e = 1; e < NEXP; ++e) if (acc[e] > v0) { v0 = acc[e]; i0 = e; }
    int i1 = (i0 == 0) ? 1 : 0; float v1 = acc[i1];
#pragma unroll
    for (int e = 0; e < NEXP; ++e)
      if (e != i0 && acc[e] > v1) { v1 = acc[e]; i1 = e; }
    float g0 = 1.f / (1.f + __expf(v1 - v0));
    topi[t * 2] = i0; topi[t * 2 + 1] = i1;
    gates[t * 2] = g0; gates[t * 2 + 1] = 1.f - g0;
    atomicAdd(&counts[i0], 1);
    atomicAdd(&counts[i1], 1);
  }
}

__global__ void scan_kernel(const int* __restrict__ counts,
                            int* __restrict__ offs, int* __restrict__ cursor) {
  if (threadIdx.x == 0 && blockIdx.x == 0) {
    int s = 0;
    for (int e = 0; e < NEXP; ++e) { offs[e] = s; s += counts[e]; cursor[e] = 0; }
    offs[NEXP] = s;
  }
}

__global__ __launch_bounds__(256) void assign_kernel(
    const int* __restrict__ topi, int* __restrict__ cursor,
    const int* __restrict__ offs, int* __restrict__ row_token,
    int* __restrict__ token_rows) {
  int t = blockIdx.x * 256 + threadIdx.x;
#pragma unroll
  for (int kk = 0; kk < 2; ++kk) {
    int e = topi[t * 2 + kk];
    int pos = atomicAdd(&cursor[e], 1);
    int row = offs[e] + pos;
    row_token[row] = t;
    token_rows[t * 2 + kk] = row;
  }
}

// ---- Expert GEMM 1: act[r] = gelu(h2[token(r)] @ W1[e]), gathered rows ----
__global__ __launch_bounds__(256) void moe_gemm1_gelu(
    const bf16_t* __restrict__ h2, const bf16_t* __restrict__ W1T,
    bf16_t* __restrict__ act, const int* __restrict__ offs,
    const int* __restrict__ row_token) {
  int e = blockIdx.z;
  int rbeg = offs[e], rend = offs[e + 1];
  int tid = threadIdx.x, lane = tid & 31, wave = tid >> 5;
  int wm = wave & 3, wn = wave >> 2;
  int m0 = rbeg + blockIdx.y * 64 + wm * 16;
  if (m0 >= rend) return;                            // wave-uniform
  int n0 = blockIdx.x * 128 + wn * 64;
  int r = imin(m0 + (lane & 15), rend - 1);
  int tok = row_token[r];
  const bf16_t* Wt = W1T + (size_t)e * HID * DIM;    // [HID x DIM]
  v8f acc[4] = {};
  for (int k = 0; k < DIM; k += 32) {
    v16bf a = frag_a(h2, DIM, tok, k, lane);
#pragma unroll
    for (int j = 0; j < 4; ++j) {
      v16bf b = frag_b(Wt, DIM, n0 + j * 16 + (lane & 15), k, lane);
      acc[j] = wmma_bf16(a, b, acc[j]);
    }
  }
#pragma unroll
  for (int j = 0; j < 4; ++j) {
    int col = n0 + j * 16 + (lane & 15);
#pragma unroll
    for (int rr = 0; rr < 8; ++rr) {
      int row = m0 + rr + ((lane & 16) ? 8 : 0);
      if (row < rend)
        act[(size_t)row * HID + col] = (bf16_t)gelu_f(acc[j][rr]);
    }
  }
}

// ---- Expert GEMM 2: C[r] = act[r] @ W2[e] (f32 out) -----------------------
__global__ __launch_bounds__(256) void moe_gemm2(
    const bf16_t* __restrict__ act, const bf16_t* __restrict__ W2T,
    float* __restrict__ C, const int* __restrict__ offs) {
  int e = blockIdx.z;
  int rbeg = offs[e], rend = offs[e + 1];
  int tid = threadIdx.x, lane = tid & 31, wave = tid >> 5;
  int wm = wave & 3, wn = wave >> 2;
  int m0 = rbeg + blockIdx.y * 64 + wm * 16;
  if (m0 >= rend) return;
  int n0 = blockIdx.x * 128 + wn * 64;
  int rowA = imin(m0 + (lane & 15), rend - 1);
  const bf16_t* Wt = W2T + (size_t)e * DIM * HID;    // [DIM x HID]
  v8f acc[4] = {};
  for (int k = 0; k < HID; k += 32) {
    v16bf a = frag_a(act, HID, rowA, k, lane);
#pragma unroll
    for (int j = 0; j < 4; ++j) {
      v16bf b = frag_b(Wt, HID, n0 + j * 16 + (lane & 15), k, lane);
      acc[j] = wmma_bf16(a, b, acc[j]);
    }
  }
#pragma unroll
  for (int j = 0; j < 4; ++j) {
    int col = n0 + j * 16 + (lane & 15);
#pragma unroll
    for (int rr = 0; rr < 8; ++rr) {
      int row = m0 + rr + ((lane & 16) ? 8 : 0);
      if (row < rend) C[(size_t)row * DIM + col] = acc[j][rr];
    }
  }
}

// ---- Final combine: out = x_orig + g0*C[row0] + g1*C[row1] ----------------
__global__ __launch_bounds__(256) void combine_kernel(
    const float* __restrict__ skip, const float* __restrict__ C,
    const int* __restrict__ token_rows, const float* __restrict__ gates,
    float* __restrict__ out) {
  size_t idx = (size_t)blockIdx.x * 256 + threadIdx.x;
  int t = (int)(idx >> 10);
  int d = (int)(idx & 1023);
  int r0 = token_rows[t * 2], r1 = token_rows[t * 2 + 1];
  float g0 = gates[t * 2], g1 = gates[t * 2 + 1];
  out[idx] = skip[idx] + g0 * C[(size_t)r0 * DIM + d]
                       + g1 * C[(size_t)r1 * DIM + d];
}

// ============================================================================
extern "C" void kernel_launch(void* const* d_in, const int* in_sizes, int n_in,
                              void* d_out, int out_size, void* d_ws,
                              size_t ws_size, hipStream_t stream) {
  (void)in_sizes; (void)n_in; (void)out_size; (void)ws_size;
  const float* x  = (const float*)d_in[0];
  const float* Wq = (const float*)d_in[1];
  const float* Wk = (const float*)d_in[2];
  const float* Wv = (const float*)d_in[3];
  const float* Wo = (const float*)d_in[4];
  const float* Wg = (const float*)d_in[5];
  const float* W1 = (const float*)d_in[6];
  const float* W2 = (const float*)d_in[7];
  float* out = (float*)d_out;

  // ---- workspace layout (~283 MB) ----
  char* ws = (char*)d_ws;
  size_t off = 0;
  auto alloc = [&](size_t bytes) -> char* {
    char* p = ws + off;
    off = (off + bytes + 255) & ~(size_t)255;
    return p;
  };
  bf16_t* hB   = (bf16_t*)alloc((size_t)N_TOK * DIM * 2);
  bf16_t* qB   = (bf16_t*)alloc((size_t)N_TOK * DIM * 2);
  bf16_t* kB   = (bf16_t*)alloc((size_t)N_TOK * HDIM * 2);
  bf16_t* vTB  = (bf16_t*)alloc((size_t)HDIM * N_TOK * 2);
  bf16_t* oB   = (bf16_t*)alloc((size_t)N_TOK * DIM * 2);
  bf16_t* x1B  = (bf16_t*)alloc((size_t)N_TOK * DIM * 2);
  bf16_t* h2B  = (bf16_t*)alloc((size_t)N_TOK * DIM * 2);
  bf16_t* WqT  = (bf16_t*)alloc((size_t)DIM * DIM * 2);
  bf16_t* WkT  = (bf16_t*)alloc((size_t)HDIM * DIM * 2);
  bf16_t* WvT  = (bf16_t*)alloc((size_t)HDIM * DIM * 2);
  bf16_t* WoT  = (bf16_t*)alloc((size_t)DIM * DIM * 2);
  bf16_t* W1T  = (bf16_t*)alloc((size_t)NEXP * HID * DIM * 2);
  bf16_t* W2T  = (bf16_t*)alloc((size_t)NEXP * DIM * HID * 2);
  bf16_t* actB = (bf16_t*)alloc((size_t)RMAX * HID * 2);
  float*  Cbuf = (float*)alloc((size_t)RMAX * DIM * 4);
  int*   counts = (int*)alloc(NEXP * 4);
  int*   offsE  = (int*)alloc((NEXP + 1) * 4);
  int*   cursor = (int*)alloc(NEXP * 4);
  int*   topi   = (int*)alloc((size_t)N_TOK * 2 * 4);
  float* gates  = (float*)alloc((size_t)N_TOK * 2 * 4);
  int*   rowTok = (int*)alloc(RMAX * 4);
  int*   tokRow = (int*)alloc((size_t)N_TOK * 2 * 4);

  // 1. h = rmsnorm(x)
  rmsnorm_kernel<float><<<N_TOK, 256, 0, stream>>>(x, hB);

  // 2. transpose+cvt weights -> bf16 BT layout
  transpose_cvt_kernel<<<dim3(32, 32, 1), 256, 0, stream>>>(Wq, WqT, DIM, DIM);
  transpose_cvt_kernel<<<dim3(2, 32, 1), 256, 0, stream>>>(Wk, WkT, DIM, HDIM);
  transpose_cvt_kernel<<<dim3(2, 32, 1), 256, 0, stream>>>(Wv, WvT, DIM, HDIM);
  transpose_cvt_kernel<<<dim3(32, 32, 1), 256, 0, stream>>>(Wo, WoT, DIM, DIM);
  transpose_cvt_kernel<<<dim3(128, 32, NEXP), 256, 0, stream>>>(W1, W1T, DIM, HID);
  transpose_cvt_kernel<<<dim3(32, 128, NEXP), 256, 0, stream>>>(W2, W2T, HID, DIM);

  // 3. q (scaled by HD^-0.5 = 1/8), k, vT
  gemm_bf16_wmma<EPI_SCALE_BF16><<<dim3(8, 64), 256, 0, stream>>>(
      hB, DIM, WqT, DIM, qB, DIM, N_TOK, DIM, DIM, nullptr, 0.125f);
  gemm_bf16_wmma<EPI_BF16><<<dim3(1, 64), 256, 0, stream>>>(
      hB, DIM, WkT, DIM, kB, HDIM, N_TOK, HDIM, DIM, nullptr, 1.f);
  gemm_bf16_wmma<EPI_TRANS_BF16><<<dim3(1, 64), 256, 0, stream>>>(
      hB, DIM, WvT, DIM, vTB, N_TOK, N_TOK, HDIM, DIM, nullptr, 1.f);

  // 4. flash attention -> o
  attn_kernel<<<dim3(SEQ / 64, NHEAD, 2), 128, 0, stream>>>(qB, kB, vTB, oB);

  // 5. x1 = o@Wo + h
  gemm_bf16_wmma<EPI_ADD_BF16><<<dim3(8, 64), 256, 0, stream>>>(
      oB, DIM, WoT, DIM, x1B, DIM, N_TOK, DIM, DIM, hB, 1.f);

  // 6. h2 = rmsnorm(x1)
  rmsnorm_kernel<bf16_t><<<N_TOK, 256, 0, stream>>>(x1B, h2B);

  // 7. gating + routing
  hipMemsetAsync(counts, 0, NEXP * sizeof(int), stream);
  gate_topk_kernel<<<N_TOK / 8, 256, 0, stream>>>(h2B, Wg, topi, gates, counts);
  scan_kernel<<<1, 1, 0, stream>>>(counts, offsE, cursor);
  assign_kernel<<<N_TOK / 256, 256, 0, stream>>>(topi, cursor, offsE, rowTok,
                                                 tokRow);

  // 8/9. grouped expert GEMMs (early-exit blocks beyond each expert's rows)
  moe_gemm1_gelu<<<dim3(HID / 128, 64, NEXP), 256, 0, stream>>>(
      h2B, W1T, actB, offsE, rowTok);
  moe_gemm2<<<dim3(DIM / 128, 64, NEXP), 256, 0, stream>>>(
      actB, W2T, Cbuf, offsE);

  // 10. out = skip + g0*C0 + g1*C1
  combine_kernel<<<(N_TOK * DIM) / 256, 256, 0, stream>>>(x, Cbuf, tokRow,
                                                          gates, out);
}
